// MoELoRALayer_9852654977364
// MI455X (gfx1250) — matrix operand order repaired and checked by reading, not compiled, not run on hardware
//
#include <hip/hip_runtime.h>

typedef __bf16 bf16_t;
typedef __attribute__((ext_vector_type(8)))  __bf16 v8bf;
typedef __attribute__((ext_vector_type(16))) __bf16 v16bf;
typedef __attribute__((ext_vector_type(8)))  float  v8f;

#define N_TOK   8192
#define D_IN    4096
#define D_OUT   4096
#define NEXP    8
#define RANK    16
#define ER      128        // NEXP*RANK
#define SCALING 2.0f

#define BT  128            // block tile: tokens
#define BO  256            // block tile: outputs (main GEMM)
#define BK  32             // K step (one bf16 WMMA)
#define LDP 40             // padded LDS row (elements) -> 80B rows, 16B aligned chunks

// ---------------- fp32 -> bf16 convert ----------------
__global__ void cvt_f32_to_bf16(const float* __restrict__ in,
                                bf16_t* __restrict__ out, int n) {
  int i = (blockIdx.x * blockDim.x + threadIdx.x) * 4;
  if (i + 3 < n) {
    const float4 f = *(const float4*)(in + i);
    out[i + 0] = (bf16_t)f.x;
    out[i + 1] = (bf16_t)f.y;
    out[i + 2] = (bf16_t)f.z;
    out[i + 3] = (bf16_t)f.w;
  } else {
    for (; i < n; ++i) out[i] = (bf16_t)in[i];
  }
}

// ---------------- B[e, o, r] -> Bcat[o, e*16+r] (bf16) ----------------
__global__ void build_bcat(const float* __restrict__ B, bf16_t* __restrict__ bc) {
  int idx = blockIdx.x * blockDim.x + threadIdx.x;
  if (idx >= D_OUT * ER) return;
  int o = idx >> 7;
  int j = idx & 127;
  int e = j >> 4;
  int r = j & 15;
  bc[idx] = (bf16_t)B[((size_t)e * D_OUT + o) * RANK + r];
}

// ---------------- router: logits -> top2 -> renormalized combine ----------------
__global__ __launch_bounds__(256)
void router_kernel(const bf16_t* __restrict__ xb,
                   const float* __restrict__ Wr,
                   float* __restrict__ comb) {
  const int n    = blockIdx.x;
  const int wave = threadIdx.x >> 5;   // 8 waves = 8 experts
  const int lane = threadIdx.x & 31;
  __shared__ float sLog[NEXP];

  const bf16_t* xr = xb + (size_t)n * D_IN;
  const float*  wr = Wr + (size_t)wave * D_IN;
  float acc = 0.f;
  for (int k = lane; k < D_IN; k += 32)
    acc += (float)xr[k] * wr[k];
#pragma unroll
  for (int off = 16; off > 0; off >>= 1)
    acc += __shfl_xor(acc, off, 32);
  if (lane == 0) sLog[wave] = acc;
  __syncthreads();
  if (threadIdx.x == 0) {
    int i1 = 0;
#pragma unroll
    for (int e = 1; e < NEXP; ++e) if (sLog[e] > sLog[i1]) i1 = e;
    int i2 = (i1 == 0) ? 1 : 0;
#pragma unroll
    for (int e = 0; e < NEXP; ++e)
      if (e != i1 && sLog[e] > sLog[i2]) i2 = e;
    // softmax denominator cancels under top-2 renormalization:
    const float w1 = 1.f / (1.f + __expf(sLog[i2] - sLog[i1]));
    const float w2 = 1.f - w1;
#pragma unroll
    for (int e = 0; e < NEXP; ++e)
      comb[(size_t)n * NEXP + e] = (e == i1) ? w1 : ((e == i2) ? w2 : 0.f);
  }
}

// ---------------- WMMA fragment loads from LDS ----------------
// A 16x32 bf16: lane (r,h): elems 0..7 -> K = h*8+0..7 ; elems 8..15 -> K = 16+h*8+0..7
__device__ __forceinline__ v16bf frag_a(const bf16_t* row, int hl) {
  const v8bf lo = *(const v8bf*)(row + hl * 8);
  const v8bf hi = *(const v8bf*)(row + 16 + hl * 8);
  return __builtin_shufflevector(lo, hi, 0, 1, 2, 3, 4, 5, 6, 7,
                                 8, 9, 10, 11, 12, 13, 14, 15);
}
// B 32x16 bf16: lane (r,h): elems 0..7 -> K = h*8+0..7 ; elems 8..15 -> K = 16+h*8+0..7
__device__ __forceinline__ v16bf frag_b(const bf16_t* row, int hl) {
  const v8bf lo = *(const v8bf*)(row + hl * 8);
  const v8bf hi = *(const v8bf*)(row + 16 + hl * 8);
  return __builtin_shufflevector(lo, hi, 0, 1, 2, 3, 4, 5, 6, 7,
                                 8, 9, 10, 11, 12, 13, 14, 15);
}

#define WMMA_BF16(a, b, c) \
  __builtin_amdgcn_wmma_f32_16x16x32_bf16(false, (a), false, (b), (short)0, (c), false, false)

// ---------------- h-GEMM: hw = (x @ Acat^T) * combine * SCALING -> bf16 ----------------
__global__ __launch_bounds__(256)
void moe_h_gemm(const bf16_t* __restrict__ Xb,     // [N_TOK][D_IN]
                const bf16_t* __restrict__ Ab,     // [ER][D_IN]
                const float*  __restrict__ comb,   // [N_TOK][NEXP]
                bf16_t*       __restrict__ HWb) {  // [N_TOK][ER]
  __shared__ __align__(16) bf16_t sA[BT][LDP];
  __shared__ __align__(16) bf16_t sB[ER][LDP];
  __shared__ float sC[BT][NEXP];

  const int tb   = blockIdx.x * BT;
  const int t    = threadIdx.x;
  const int wave = t >> 5;
  const int lane = t & 31;
  const int wm = wave & 1, wn = wave >> 1;
  const int r = lane & 15, hl = lane >> 4;

  for (int idx = t; idx < BT * NEXP; idx += 256)
    sC[idx >> 3][idx & 7] = comb[(size_t)(tb + (idx >> 3)) * NEXP + (idx & 7)];

  v8f acc[4][2] = {};

  for (int k0 = 0; k0 < D_IN; k0 += BK) {
#pragma unroll
    for (int i = 0; i < 2; ++i) {
      const int c = t + i * 256;
      const int row = c >> 2;
      const int col = (c & 3) << 3;
      *(v8bf*)&sA[row][col] = *(const v8bf*)(Xb + (size_t)(tb + row) * D_IN + k0 + col);
      *(v8bf*)&sB[row][col] = *(const v8bf*)(Ab + (size_t)row * D_IN + k0 + col);
    }
    __syncthreads();
    const v16bf b0 = frag_b(&sB[wn * 32 + r][0], hl);
    const v16bf b1 = frag_b(&sB[wn * 32 + 16 + r][0], hl);
#pragma unroll
    for (int i = 0; i < 4; ++i) {
      const v16bf a = frag_a(&sA[wm * 64 + i * 16 + r][0], hl);
      acc[i][0] = WMMA_BF16(a, b0, acc[i][0]);
      acc[i][1] = WMMA_BF16(a, b1, acc[i][1]);
    }
    __syncthreads();
  }

#pragma unroll
  for (int j = 0; j < 2; ++j) {
    const int col = wn * 32 + j * 16 + r;     // 0..127
    const int e = col >> 4;
#pragma unroll
    for (int i = 0; i < 4; ++i) {
      const int rl0 = wm * 64 + i * 16 + hl * 8;
#pragma unroll
      for (int v = 0; v < 8; ++v) {
        const float val = acc[i][j][v] * sC[rl0 + v][e] * SCALING;
        HWb[(size_t)(tb + rl0 + v) * ER + col] = (bf16_t)val;
      }
    }
  }
}

// ---------------- main GEMM: out = x@W^T + bias + hw@Bcat^T ----------------
// Block tile 128 (tokens) x 256 (outputs); 8 waves in 2x4 grid; wave tile 64x64
// => per wave per K-step: 16 WMMAs fed by 16 ds_load_b128 (1.0 LDS frag load / WMMA).
__global__ __launch_bounds__(256)
void moe_main_gemm(const bf16_t* __restrict__ Xb,    // [N_TOK][D_IN]
                   const bf16_t* __restrict__ Wb,    // [D_OUT][D_IN]
                   const bf16_t* __restrict__ HWb,   // [N_TOK][ER]
                   const bf16_t* __restrict__ BCb,   // [D_OUT][ER]
                   const float*  __restrict__ bias,  // [D_OUT]
                   float*        __restrict__ Out) { // [N_TOK][D_OUT]
  __shared__ __align__(16) bf16_t sA[BT][LDP];
  __shared__ __align__(16) bf16_t sB[BO][LDP];

  const int tb = blockIdx.y * BT;
  const int ob = blockIdx.x * BO;
  const int t    = threadIdx.x;
  const int wave = t >> 5;
  const int lane = t & 31;
  const int wm = wave & 1;        // 0..1 -> 64-row slice of tokens
  const int wn = wave >> 1;       // 0..3 -> 64-col slice of outputs
  const int r = lane & 15, hl = lane >> 4;

  v8f acc[4][4] = {};

  // ---- main K loop over d_in ----
  for (int k0 = 0; k0 < D_IN; k0 += BK) {
#pragma unroll
    for (int i = 0; i < 2; ++i) {
      const int c = t + i * 256;
      const int row = c >> 2;
      const int col = (c & 3) << 3;
      *(v8bf*)&sA[row][col] = *(const v8bf*)(Xb + (size_t)(tb + row) * D_IN + k0 + col);
    }
#pragma unroll
    for (int i = 0; i < 4; ++i) {
      const int c = t + i * 256;
      const int row = c >> 2;
      const int col = (c & 3) << 3;
      *(v8bf*)&sB[row][col] = *(const v8bf*)(Wb + (size_t)(ob + row) * D_IN + k0 + col);
    }
    __syncthreads();
    v16bf bfr[4];
#pragma unroll
    for (int j = 0; j < 4; ++j)
      bfr[j] = frag_b(&sB[wn * 64 + j * 16 + r][0], hl);
#pragma unroll
    for (int i = 0; i < 4; ++i) {
      const v16bf a = frag_a(&sA[wm * 64 + i * 16 + r][0], hl);
#pragma unroll
      for (int j = 0; j < 4; ++j)
        acc[i][j] = WMMA_BF16(a, bfr[j], acc[i][j]);
    }
    __syncthreads();
  }

  // ---- expert tail: K = ER = 128 over hw @ Bcat^T ----
  for (int k0 = 0; k0 < ER; k0 += BK) {
#pragma unroll
    for (int i = 0; i < 2; ++i) {
      const int c = t + i * 256;
      const int row = c >> 2;
      const int col = (c & 3) << 3;
      *(v8bf*)&sA[row][col] = *(const v8bf*)(HWb + (size_t)(tb + row) * ER + k0 + col);
    }
#pragma unroll
    for (int i = 0; i < 4; ++i) {
      const int c = t + i * 256;
      const int row = c >> 2;
      const int col = (c & 3) << 3;
      *(v8bf*)&sB[row][col] = *(const v8bf*)(BCb + (size_t)(ob + row) * ER + k0 + col);
    }
    __syncthreads();
    v16bf bfr[4];
#pragma unroll
    for (int j = 0; j < 4; ++j)
      bfr[j] = frag_b(&sB[wn * 64 + j * 16 + r][0], hl);
#pragma unroll
    for (int i = 0; i < 4; ++i) {
      const v16bf a = frag_a(&sA[wm * 64 + i * 16 + r][0], hl);
#pragma unroll
      for (int j = 0; j < 4; ++j)
        acc[i][j] = WMMA_BF16(a, bfr[j], acc[i][j]);
    }
    __syncthreads();
  }

  // ---- epilogue: + bias, store fp32 ----
#pragma unroll
  for (int j = 0; j < 4; ++j) {
    const int col = ob + wn * 64 + j * 16 + r;
    const float bv = bias[col];
#pragma unroll
    for (int i = 0; i < 4; ++i) {
      const int row0 = tb + wm * 64 + i * 16 + hl * 8;
#pragma unroll
      for (int v = 0; v < 8; ++v)
        Out[(size_t)(row0 + v) * D_OUT + col] = acc[i][j][v] + bv;
    }
  }
}

// ---------------- host launcher ----------------
extern "C" void kernel_launch(void* const* d_in, const int* in_sizes, int n_in,
                              void* d_out, int out_size, void* d_ws, size_t ws_size,
                              hipStream_t stream) {
  const float* x        = (const float*)d_in[0];  // [4,2048,4096]
  const float* W_base   = (const float*)d_in[1];  // [4096,4096]
  const float* b_base   = (const float*)d_in[2];  // [4096]
  const float* W_router = (const float*)d_in[3];  // [8,4096]
  const float* A        = (const float*)d_in[4];  // [8,16,4096] == Acat[128][4096]
  const float* B        = (const float*)d_in[5];  // [8,4096,16]
  float* out = (float*)d_out;

  char* ws = (char*)d_ws;
  size_t off = 0;
  auto alloc = [&](size_t bytes) -> void* {
    void* p = ws + off;
    off += (bytes + 255) & ~(size_t)255;
    return p;
  };
  bf16_t* xb   = (bf16_t*)alloc((size_t)N_TOK * D_IN * 2);   // 64 MB
  bf16_t* wb   = (bf16_t*)alloc((size_t)D_OUT * D_IN * 2);   // 32 MB
  bf16_t* ab   = (bf16_t*)alloc((size_t)ER * D_IN * 2);      // 1 MB
  bf16_t* bcb  = (bf16_t*)alloc((size_t)D_OUT * ER * 2);     // 1 MB
  bf16_t* hwb  = (bf16_t*)alloc((size_t)N_TOK * ER * 2);     // 2 MB
  float*  comb = (float*)alloc((size_t)N_TOK * NEXP * 4);    // 256 KB

  const int nx = N_TOK * D_IN;
  cvt_f32_to_bf16<<<(nx / 4 + 255) / 256, 256, 0, stream>>>(x, xb, nx);
  const int nw = D_OUT * D_IN;
  cvt_f32_to_bf16<<<(nw / 4 + 255) / 256, 256, 0, stream>>>(W_base, wb, nw);
  const int na = ER * D_IN;
  cvt_f32_to_bf16<<<(na / 4 + 255) / 256, 256, 0, stream>>>(A, ab, na);
  build_bcat<<<(D_OUT * ER + 255) / 256, 256, 0, stream>>>(B, bcb);

  router_kernel<<<N_TOK, 256, 0, stream>>>(xb, W_router, comb);
  moe_h_gemm<<<N_TOK / BT, 256, 0, stream>>>(xb, ab, comb, hwb);
  moe_main_gemm<<<dim3(D_OUT / BO, N_TOK / BT), 256, 0, stream>>>(
      xb, wb, hwb, bcb, b_base, out);
}
